// PaddingNet_79182017069466
// MI455X (gfx1250) — 8-bit matrix operand order repaired and checked by reading, compiled, not run on hardware
//
#include <hip/hip_runtime.h>

typedef signed char i8;
typedef __attribute__((ext_vector_type(4))) int v4i;
typedef __attribute__((ext_vector_type(8))) int v8i;

#define DEVFN static __device__ __forceinline__

#if __has_builtin(__builtin_amdgcn_global_load_async_to_lds_b128) && \
    __has_builtin(__builtin_amdgcn_s_wait_asynccnt)
#define USE_ASYNC_LDS 1
#else
#define USE_ASYNC_LDS 0
#endif

DEVFN float qclip(float v) { return fminf(fmaxf(v, -128.f), 127.f); }

DEVFN i8 relu_fq_chain(float y, float sr, float sq) {
    y = fmaxf(y, 0.f);
    float v1 = qclip(rintf(y / sr)) * sr;
    float q2 = qclip(rintf(v1 / sq));
    return (i8)(int)q2;
}

#if USE_ASYNC_LDS
DEVFN void async_copy16(const void* g, void* l) {
    __builtin_amdgcn_global_load_async_to_lds_b128((v4i*)g, (v4i*)l, 0, 0);
}
#endif

__global__ __launch_bounds__(256) void quant_input_kernel(
    const float* __restrict__ x, const float* __restrict__ s_p,
    i8* __restrict__ q, int n) {
    int i = blockIdx.x * 256 + threadIdx.x;
    if (i < n) {
        float s = *s_p;
        q[i] = (i8)(int)qclip(rintf(x[i] / s));
    }
}

__global__ __launch_bounds__(256) void prep_flat_weight_kernel(
    const float* __restrict__ W, int n, float* __restrict__ scale_out,
    i8* __restrict__ Q) {
    __shared__ float red[256];
    int tid = threadIdx.x;
    float mx = 0.f;
    for (int i = tid; i < n; i += 256) mx = fmaxf(mx, fabsf(W[i]));
    red[tid] = mx;
    __syncthreads();
    for (int s = 128; s > 0; s >>= 1) {
        if (tid < s) red[tid] = fmaxf(red[tid], red[tid + s]);
        __syncthreads();
    }
    float sc = red[0] / 127.f;
    if (tid == 0) *scale_out = sc;
    for (int i = tid; i < n; i += 256)
        Q[i] = (i8)(int)qclip(rintf(W[i] / sc));
}

template <int COUT, int CIN, int KH, int KW, int KSTEPS, int NTILES>
__global__ __launch_bounds__(256) void prep_wmma_weight_kernel(
    const float* __restrict__ W, float* __restrict__ scale_out,
    int* __restrict__ Bpk) {
    __shared__ float red[256];
    const int KDIM = CIN * KH * KW;
    const int NW = COUT * KDIM;
    int tid = threadIdx.x;
    float mx = 0.f;
    for (int i = tid; i < NW; i += 256) mx = fmaxf(mx, fabsf(W[i]));
    red[tid] = mx;
    __syncthreads();
    for (int s = 128; s > 0; s >>= 1) {
        if (tid < s) red[tid] = fmaxf(red[tid], red[tid + s]);
        __syncthreads();
    }
    float sc = red[0] / 127.f;
    if (tid == 0) *scale_out = sc;

    const int TOT = KSTEPS * NTILES * 32 * 8;
    for (int d = tid; d < TOT; d += 256) {
        int t = d / (NTILES * 256);
        int rem = d % (NTILES * 256);
        int nt = rem / 256; rem %= 256;
        int lane = rem >> 3;
        int v = rem & 7;
        int N = nt * 16 + (lane & 15);
        int kb = t * 64 + ((v >> 2) * 32) + ((lane >> 4) * 16) + (v & 3) * 4;
        unsigned dw = 0;
        #pragma unroll
        for (int j = 0; j < 4; ++j) {
            int k = kb + j;
            int q = 0;
            if (k < KDIM && N < COUT) {
                int ci = k % CIN;
                int pos = k / CIN;
                int kx = pos % KW;
                int ky = pos / KW;
                float w = W[((N * CIN + ci) * KH + ky) * KW + kx];
                q = (int)qclip(rintf(w / sc));
            }
            dw |= ((unsigned)(q & 0xff)) << (8 * j);
        }
        Bpk[d] = (int)dw;
    }
}

__global__ __launch_bounds__(256) void conv1_kernel(
    const i8* __restrict__ qin, const i8* __restrict__ qw,
    const float* __restrict__ sw_p, const float* __restrict__ bias,
    const float* __restrict__ sa_p, const float* __restrict__ sr_p,
    const float* __restrict__ sq_p, i8* __restrict__ out) {
    int idx = blockIdx.x * 256 + threadIdx.x;
    if (idx >= 32 * 254 * 254 * 16) return;
    int co = idx & 15;
    int r = idx >> 4;
    int ox = r % 254; r /= 254;
    int oy = r % 254;
    int n = r / 254;

    const i8* ip = qin + (n * 256 + oy) * 256 + ox;
    const i8* wp = qw + co * 9;
    int acc = 0;
    #pragma unroll
    for (int ky = 0; ky < 3; ++ky)
        #pragma unroll
        for (int kx = 0; kx < 3; ++kx)
            acc += (int)ip[ky * 256 + kx] * (int)wp[ky * 3 + kx];

    float y = (float)acc * ((*sa_p) * (*sw_p)) + bias[co];
    out[idx] = relu_fq_chain(y, *sr_p, *sq_p);
}

template <int CIN, int KH, int KW, int PAD, int COUT, int KSTEPS, int NTILES,
          int HIN, int WIN, int HOUT, int WOUT>
__global__ __launch_bounds__(256) void conv_wmma_kernel(
    const i8* __restrict__ in, const int* __restrict__ Bpk,
    const float* __restrict__ sw_p, const float* __restrict__ bias,
    const float* __restrict__ sa_p, const float* __restrict__ sr_p,
    const float* __restrict__ sq_p, i8* __restrict__ out) {
    constexpr int KDIM = KH * KW * CIN;
    constexpr int HW = HOUT * WOUT;
    constexpr int NTOT = 32 * HW;
    constexpr int TILES_M = (NTOT + 15) / 16;
    constexpr int BDW = KSTEPS * NTILES * 256;

    __shared__ __align__(32) int Blds[BDW];

#if USE_ASYNC_LDS
    {
        constexpr int CHUNKS = BDW / 4;
        for (int i = threadIdx.x; i < CHUNKS; i += 256)
            async_copy16((const char*)Bpk + i * 16, (char*)Blds + i * 16);
        __builtin_amdgcn_s_wait_asynccnt(0);
    }
#else
    for (int i = threadIdx.x; i < BDW; i += 256) Blds[i] = Bpk[i];
#endif
    __syncthreads();

    const int mtile = blockIdx.x * 8 + (threadIdx.x >> 5);
    const int lane = threadIdx.x & 31;
    if (mtile < TILES_M) {
        const int m = mtile * 16 + (lane & 15);
        const int h = lane >> 4;
        const bool mvalid = m < NTOT;
        int n = 0, oy = 0, ox = 0;
        if (mvalid) {
            n = m / HW;
            const int r = m % HW;
            oy = r / WOUT;
            ox = r % WOUT;
        }
        const int nbase = n * (HIN * WIN * CIN);

        v8i accs[NTILES];
        #pragma unroll
        for (int nt = 0; nt < NTILES; ++nt) accs[nt] = (v8i){};

        #pragma unroll
        for (int t = 0; t < KSTEPS; ++t) {
            v8i a;
            #pragma unroll
            for (int v = 0; v < 8; ++v) {
                const int kb = t * 64 + (v >> 2) * 32 + h * 16 + (v & 3) * 4;
                int q = 0;
                if (kb < KDIM) {
                    const int ci = kb % CIN;
                    const int pos = kb / CIN;
                    const int ky = pos / KW;
                    const int kx = pos % KW;
                    const int iy = oy + ky - PAD;
                    const int ix = ox + kx - PAD;
                    if (mvalid && (unsigned)iy < (unsigned)HIN &&
                        (unsigned)ix < (unsigned)WIN)
                        q = *(const int*)(in + nbase +
                                          ((iy * WIN + ix) * CIN + ci));
                }
                a[v] = q;
            }
            #pragma unroll
            for (int nt = 0; nt < NTILES; ++nt) {
                const v8i b =
                    *(const v8i*)(Blds + ((t * NTILES + nt) * 32 + lane) * 8);
                accs[nt] = __builtin_amdgcn_wmma_i32_16x16x64_iu8(
                    true, a, true, b, accs[nt], false, false);
            }
        }

        const float scale = (*sa_p) * (*sw_p);
        const float sr = *sr_p, sq = *sq_p;
        #pragma unroll
        for (int nt = 0; nt < NTILES; ++nt) {
            const int nn = nt * 16 + (lane & 15);
            #pragma unroll
            for (int r = 0; r < 8; ++r) {
                const int p = mtile * 16 + r + h * 8;
                if (p < NTOT && nn < COUT) {
                    const float y = (float)accs[nt][r] * scale + bias[nn];
                    out[p * COUT + nn] = relu_fq_chain(y, sr, sq);
                }
            }
        }
    }
}

__global__ __launch_bounds__(256) void maxpool_quant_kernel(
    const i8* __restrict__ in, const float* __restrict__ sin_p,
    const float* __restrict__ sout_p, i8* __restrict__ out) {
    int idx = blockIdx.x * 256 + threadIdx.x;
    if (idx >= 32 * 127 * 127 * 32) return;
    int c = idx & 31;
    int r = idx >> 5;
    int ox = r % 127; r /= 127;
    int oy = r % 127;
    int n = r / 127;
    const i8* ip = in + ((n * 254 + oy * 2) * 254 + ox * 2) * 32 + c;
    constexpr int RS = 254 * 32;
    int m = (int)ip[0];
    m = max(m, (int)ip[32]);
    m = max(m, (int)ip[RS]);
    m = max(m, (int)ip[RS + 32]);
    float v = (float)m * (*sin_p);
    out[idx] = (i8)(int)qclip(rintf(v / (*sout_p)));
}

__global__ __launch_bounds__(256) void gap_quant_kernel(
    const i8* __restrict__ in, const float* __restrict__ sin_p,
    const float* __restrict__ sq_p, float* __restrict__ gap) {
    __shared__ int red[256];
    const int b = blockIdx.x;
    const int n = b >> 4, c = b & 15;
    const int tid = threadIdx.x;
    const i8* p = in + n * (125 * 125 * 16) + c;
    int s = 0;
    for (int i = tid; i < 15625; i += 256) s += (int)p[i * 16];
    red[tid] = s;
    __syncthreads();
    for (int k = 128; k > 0; k >>= 1) {
        if (tid < k) red[tid] += red[tid + k];
        __syncthreads();
    }
    if (tid == 0) {
        float mean = (float)red[0] * (*sin_p) / 15625.f;
        float sq = *sq_p;
        gap[b] = qclip(rintf(mean / sq)) * sq;
    }
}

__global__ __launch_bounds__(320) void classifier_kernel(
    const float* __restrict__ gap, const i8* __restrict__ qwc,
    const float* __restrict__ swc_p, const float* __restrict__ bc,
    float* __restrict__ out) {
    int i = threadIdx.x;
    if (i < 320) {
        int b = i / 10, o = i % 10;
        float swc = *swc_p;
        float acc = bc[o];
        #pragma unroll
        for (int c = 0; c < 16; ++c)
            acc += gap[b * 16 + c] * (float)qwc[o * 16 + c] * swc;
        out[b * 10 + o] = acc;
    }
}

static inline size_t align256(size_t x) { return (x + 255) & ~(size_t)255; }

extern "C" void kernel_launch(void* const* d_in, const int* in_sizes, int n_in,
                              void* d_out, int out_size, void* d_ws,
                              size_t ws_size, hipStream_t stream) {
    (void)in_sizes; (void)n_in; (void)out_size; (void)ws_size;
    const float* x  = (const float*)d_in[0];
    const float* w1 = (const float*)d_in[1];
    const float* b1 = (const float*)d_in[2];
    const float* w2 = (const float*)d_in[3];
    const float* b2 = (const float*)d_in[4];
    const float* w3 = (const float*)d_in[5];
    const float* b3 = (const float*)d_in[6];
    const float* w4 = (const float*)d_in[7];
    const float* b4 = (const float*)d_in[8];
    const float* wc = (const float*)d_in[9];
    const float* bc = (const float*)d_in[10];
    const float* s  = (const float*)d_in[11];
    float* out = (float*)d_out;

    char* ws = (char*)d_ws;
    size_t o = 0;
    auto take = [&](size_t bytes) { size_t r = o; o = align256(o + bytes); return r; };
    i8*    q_in = (i8*)(ws + take(32u * 256 * 256));
    i8*    a1   = (i8*)(ws + take(32u * 254 * 254 * 16));
    i8*    a2   = (i8*)(ws + take(32u * 254 * 254 * 32));
    i8*    a3   = (i8*)(ws + take(32u * 254 * 254 * 32));
    i8*    p1   = (i8*)(ws + take(32u * 127 * 127 * 32));
    i8*    a4   = (i8*)(ws + take(32u * 125 * 125 * 16));
    float* gap  = (float*)(ws + take(512 * sizeof(float)));
    float* wsc  = (float*)(ws + take(8 * sizeof(float)));
    int*   Bpk2 = (int*)(ws + take(3u * 2 * 32 * 8 * 4));
    int*   Bpk3 = (int*)(ws + take(13u * 2 * 32 * 8 * 4));
    int*   Bpk4 = (int*)(ws + take(5u * 1 * 32 * 8 * 4));
    i8*    qw1  = (i8*)(ws + take(256));
    i8*    qwc  = (i8*)(ws + take(256));
    float* sw1 = wsc + 0, *sw2 = wsc + 1, *sw3 = wsc + 2, *sw4 = wsc + 3,
         * swc = wsc + 4;

    prep_flat_weight_kernel<<<1, 256, 0, stream>>>(w1, 16 * 9, sw1, qw1);
    prep_wmma_weight_kernel<32, 16, 3, 3, 3, 2>
        <<<1, 256, 0, stream>>>(w2, sw2, Bpk2);
    prep_wmma_weight_kernel<32, 32, 5, 5, 13, 2>
        <<<1, 256, 0, stream>>>(w3, sw3, Bpk3);
    prep_wmma_weight_kernel<16, 32, 3, 3, 5, 1>
        <<<1, 256, 0, stream>>>(w4, sw4, Bpk4);
    prep_flat_weight_kernel<<<1, 256, 0, stream>>>(wc, 10 * 16, swc, qwc);

    quant_input_kernel<<<8192, 256, 0, stream>>>(x, s + 0, q_in,
                                                 32 * 256 * 256);

    conv1_kernel<<<129032, 256, 0, stream>>>(q_in, qw1, sw1, b1, s + 0, s + 1,
                                             s + 2, a1);

    auto blocks_for = [](int ntot) {
        int tm = (ntot + 15) / 16;
        return (tm + 7) / 8;
    };

    conv_wmma_kernel<16, 3, 3, 1, 32, 3, 2, 254, 254, 254, 254>
        <<<blocks_for(32 * 254 * 254), 256, 0, stream>>>(
            a1, Bpk2, sw2, b2, s + 2, s + 3, s + 4, a2);

    conv_wmma_kernel<32, 5, 5, 2, 32, 13, 2, 254, 254, 254, 254>
        <<<blocks_for(32 * 254 * 254), 256, 0, stream>>>(
            a2, Bpk3, sw3, b3, s + 4, s + 5, s + 6, a3);

    maxpool_quant_kernel<<<64516, 256, 0, stream>>>(a3, s + 6, s + 7, p1);

    conv_wmma_kernel<32, 3, 3, 0, 16, 5, 1, 127, 127, 125, 125>
        <<<blocks_for(32 * 125 * 125), 256, 0, stream>>>(
            p1, Bpk4, sw4, b4, s + 7, s + 8, s + 9, a4);

    gap_quant_kernel<<<512, 256, 0, stream>>>(a4, s + 9, s + 10, gap);

    classifier_kernel<<<1, 320, 0, stream>>>(gap, qwc, swc, bc, out);
}
